// DeGTAConv_83923660963943
// MI455X (gfx1250) — compile-verified, matrix operand order checked by reading, not run on hardware
//
#include <hip/hip_runtime.h>
#include <hip/hip_bf16.h>
#include <math.h>

// ---------------------------------------------------------------------------
// DeGTAConv for MI455X (gfx1250, wave32, WMMA).
// Heavy path: per-head QK^T score matrices via v_wmma_f32_16x16x32_bf16,
// two-pass (flash) row softmax, fused transposed store for column top-K.
// ---------------------------------------------------------------------------

constexpr int NN = 4096;
constexpr int EE = 65536;
constexpr int HEADS = 4;
constexpr int HDP = 32;            // padded per-head K dim for WMMA (K=32 bf16)

typedef __attribute__((ext_vector_type(16))) __bf16 v16bf;
typedef __attribute__((ext_vector_type(8)))  float  v8f;

// ------------------------------ helpers ------------------------------------

__device__ inline unsigned short f2bf(float f) {
  unsigned u = __float_as_uint(f);
  unsigned r = u + 0x7fffu + ((u >> 16) & 1u);   // RNE
  return (unsigned short)(r >> 16);
}

__device__ inline v8f wmma_bf16(v16bf a, v16bf b, v8f c) {
  // D = A(16x32 bf16) * B(32x16 bf16) + C(16x16 f32)
  return __builtin_amdgcn_wmma_f32_16x16x32_bf16(
      /*neg_a=*/false, a, /*neg_b=*/false, b,
      /*c_mod=*/(short)0, c, /*reuse_a=*/false, /*reuse_b=*/false);
}

// Load a 16x32 bf16 fragment (A layout; B of QK^T loads identically from the
// row-major query buffer). Buffer layout: row-major, 32 bf16 (=16 u32) per
// (node,head) row, stride_u32 between consecutive nodes.
// lane 0-15 : row = row0+lane,    u32 idx {0..3, 8..11}   (K 0..7,16..23)
// lane 16-31: row = row0+lane-16, u32 idx {4..7, 12..15}  (K 8..15,24..31)
__device__ inline v16bf load_frag(const unsigned* __restrict__ base,
                                  int row0, int lane, int stride_u32) {
  int l  = lane & 15;
  int hi = (lane >> 4) & 1;
  const unsigned* p = base + (size_t)(row0 + l) * stride_u32 + hi * 4;
  union { v16bf v; unsigned u[8]; } r;
  r.u[0] = p[0];  r.u[1] = p[1];  r.u[2] = p[2];  r.u[3] = p[3];
  r.u[4] = p[8];  r.u[5] = p[9];  r.u[6] = p[10]; r.u[7] = p[11];
  return r.v;
}

// order-preserving float->uint key for atomicMax-based segment max
__device__ inline unsigned fkey(float f) {
  unsigned k = __float_as_uint(f);
  return (k & 0x80000000u) ? ~k : (k | 0x80000000u);
}
__device__ inline float funkey(unsigned k) {
  return (k & 0x80000000u) ? __uint_as_float(k ^ 0x80000000u)
                           : __uint_as_float(~k);
}

// ------------------------------ encoders -----------------------------------

template <int C>
__global__ void k_mlp(const float* __restrict__ x, int xoff,
                      const float* __restrict__ w1, const float* __restrict__ b1,
                      const float* __restrict__ w2, const float* __restrict__ b2,
                      float* __restrict__ out) {
  int n = blockIdx.x, t = threadIdx.x;          // block = 2C threads
  __shared__ float xin[C];
  __shared__ float h1[2 * C];
  if (t < C) xin[t] = x[n * 96 + xoff + t];
  __syncthreads();
  float acc = b1[t];
  for (int j = 0; j < C; ++j) acc += w1[t * C + j] * xin[j];
  h1[t] = fmaxf(acc, 0.f);
  __syncthreads();
  if (t < C) {
    float o = b2[t];
    for (int j = 0; j < 2 * C; ++j) o += w2[t * 2 * C + j] * h1[j];
    out[n * C + t] = o;
  }
}

// ------------------------------ GAT ----------------------------------------

template <int C>
__global__ void k_gat_node(const float* __restrict__ h, const float* __restrict__ w,
                           const float* __restrict__ asrc, const float* __restrict__ adst,
                           float* __restrict__ ssrc, float* __restrict__ sdst) {
  int n = blockIdx.x * blockDim.x + threadIdx.x;
  if (n >= NN) return;
  float s1 = 0.f, s2 = 0.f;
  for (int i = 0; i < C; ++i) {
    float hh = 0.f;
    for (int j = 0; j < C; ++j) hh += w[i * C + j] * h[n * C + j];
    s1 += hh * asrc[i];
    s2 += hh * adst[i];
  }
  ssrc[n] = s1; sdst[n] = s2;
}

__global__ void k_gat_edge_max(const int* __restrict__ src, const int* __restrict__ dst,
                               const float* __restrict__ ssrc, const float* __restrict__ sdst,
                               float* __restrict__ tmp_a, unsigned* __restrict__ seg_m) {
  int e = blockIdx.x * blockDim.x + threadIdx.x;
  if (e >= EE) return;
  float a = ssrc[src[e]] + sdst[dst[e]];
  a = (a > 0.f) ? a : 0.2f * a;                  // leaky_relu(0.2)
  tmp_a[e] = a;
  atomicMax(&seg_m[dst[e]], fkey(a));
}

__global__ void k_gat_edge_exp(const int* __restrict__ dst,
                               float* __restrict__ tmp_a,
                               const unsigned* __restrict__ seg_m,
                               float* __restrict__ seg_s) {
  int e = blockIdx.x * blockDim.x + threadIdx.x;
  if (e >= EE) return;
  int d = dst[e];
  float ev = __expf(tmp_a[e] - funkey(seg_m[d]));
  tmp_a[e] = ev;
  atomicAdd(&seg_s[d], ev);
}

__global__ void k_gat_edge_alpha(const int* __restrict__ dst,
                                 const float* __restrict__ tmp_a,
                                 const float* __restrict__ seg_s,
                                 const float* __restrict__ coef,
                                 float* __restrict__ local_attn) {
  int e = blockIdx.x * blockDim.x + threadIdx.x;
  if (e >= EE) return;
  local_attn[e] += (*coef) * tmp_a[e] / (seg_s[dst[e]] + 1e-16f);
}

// ------------------------------ GCN local ----------------------------------

__global__ void k_gcn_h(const float* __restrict__ ae, const float* __restrict__ gw,
                        float* __restrict__ h) {
  int tid = blockIdx.x * blockDim.x + threadIdx.x;
  if (tid >= NN * 64) return;
  int n = tid >> 6, i = tid & 63;
  float acc = 0.f;
  for (int j = 0; j < 64; ++j) acc += ae[n * 64 + j] * gw[i * 64 + j];
  h[tid] = acc;
}

__global__ void k_deg_scatter(const int* __restrict__ dst, const float* __restrict__ attn,
                              float* __restrict__ deg) {
  int e = blockIdx.x * blockDim.x + threadIdx.x;
  if (e >= EE) return;
  atomicAdd(&deg[dst[e]], attn[e]);
}

__global__ void k_dinv(float* __restrict__ deg) {
  int n = blockIdx.x * blockDim.x + threadIdx.x;
  if (n >= NN) return;
  float d = deg[n];
  deg[n] = (d > 0.f) ? rsqrtf(d) : 0.f;
}

__global__ void k_local_scatter(const int* __restrict__ src, const int* __restrict__ dst,
                                const float* __restrict__ attn, const float* __restrict__ dinv,
                                const float* __restrict__ h, float* __restrict__ out_local) {
  int tid = blockIdx.x * blockDim.x + threadIdx.x;
  if (tid >= EE * 64) return;
  int e = tid >> 6, c = tid & 63;
  int s = src[e], d = dst[e];
  float w = dinv[s] * attn[e] * dinv[d];
  atomicAdd(&out_local[d * 64 + c], w * h[s * 64 + c]);
}

__global__ void k_local_self(const float* __restrict__ dinv, const float* __restrict__ h,
                             const float* __restrict__ gcn_b, float* __restrict__ out_local) {
  int tid = blockIdx.x * blockDim.x + threadIdx.x;
  if (tid >= NN * 64) return;
  int n = tid >> 6, c = tid & 63;
  out_local[tid] += dinv[n] * dinv[n] * h[tid] + gcn_b[c];
}

// ------------------------------ MHA Q/K staging -----------------------------

template <int C>
__global__ void k_qk(const float* __restrict__ feat,
                     const float* __restrict__ wq, const float* __restrict__ bq,
                     const float* __restrict__ wk, const float* __restrict__ bk,
                     float qscale,
                     unsigned short* __restrict__ qpad, unsigned short* __restrict__ kpad) {
  int tid = blockIdx.x * blockDim.x + threadIdx.x;
  if (tid >= NN * C) return;
  int n = tid / C, c = tid % C;
  const int hd = C / HEADS;
  int h = c / hd, d = c % hd;
  float q = bq[c], k = bk[c];
  for (int j = 0; j < C; ++j) {
    float xv = feat[n * C + j];
    q += xv * wq[c * C + j];
    k += xv * wk[c * C + j];
  }
  size_t o = ((size_t)(n * HEADS + h)) * HDP + d;   // pad region pre-zeroed
  qpad[o] = f2bf(q * qscale);
  kpad[o] = f2bf(k);
}

// ------------------------------ pass 1: softmax stats -----------------------
// block (32,4): wave = head; each block owns 16 query columns n.
// C tile rows = m (keys) -> reduction over m is in-lane + shfl_xor(16).

__global__ void k_mha_stats(const unsigned* __restrict__ qpad,
                            const unsigned* __restrict__ kpad,
                            float* __restrict__ stats /* [n][h][2] = {M, 1/S} */) {
  int n0 = blockIdx.x * 16;
  int h = threadIdx.y;
  int lane = threadIdx.x;
  const int stride = HEADS * (HDP / 2);   // u32 per node
  v16bf B = load_frag(qpad + h * (HDP / 2), n0, lane, stride);
  float M = -1e30f, S = 0.f;
  const unsigned* kb = kpad + h * (HDP / 2);
  for (int mt = 0; mt < NN / 16; ++mt) {
    __builtin_prefetch(kb + (size_t)((mt + 1) & (NN / 16 - 1)) * 16 * stride, 0, 1);
    v16bf A = load_frag(kb, mt * 16, lane, stride);
    v8f c = {};
    c = wmma_bf16(A, B, c);
    float tmax = c[0];
    for (int i = 1; i < 8; ++i) tmax = fmaxf(tmax, c[i]);
    tmax = fmaxf(tmax, __shfl_xor(tmax, 16, 32));
    float newM = fmaxf(M, tmax);
    float ps = 0.f;
    for (int i = 0; i < 8; ++i) ps += __expf(c[i] - newM);
    ps += __shfl_xor(ps, 16, 32);
    S = S * __expf(M - newM) + ps;
    M = newM;
  }
  if (lane < 16) {
    int n = n0 + lane;
    stats[(n * HEADS + h) * 2 + 0] = M;
    stats[(n * HEADS + h) * 2 + 1] = 1.f / S;
  }
}

// ------------------------------ pass 2: fused prob matrices -----------------
// Writes sampleT[m][n] = adjmask * (a_g*pe_g + b_g*se_g)[n][m]
// and    aegT[m][n]    = ae_g[n][m]   (transposed for column top-K)

__global__ void k_pass2(const unsigned* __restrict__ qpe, const unsigned* __restrict__ kpe,
                        const unsigned* __restrict__ qse, const unsigned* __restrict__ kse,
                        const unsigned* __restrict__ qae, const unsigned* __restrict__ kae,
                        const float* __restrict__ stpe, const float* __restrict__ stse,
                        const float* __restrict__ stae,
                        const float* __restrict__ ag, const float* __restrict__ bg,
                        const unsigned* __restrict__ bitsT,
                        float* __restrict__ sampleT, float* __restrict__ aegT) {
  int m0 = blockIdx.x * 16;
  int nt = blockIdx.y * 4 + threadIdx.y;
  int n0 = nt * 16;
  int lane = threadIdx.x;
  int ncol = n0 + (lane & 15);
  int mrow0 = m0 + ((lane >> 4) ? 8 : 0);
  const int stride = HEADS * (HDP / 2);

  const unsigned* qs[2] = {qpe, qse};
  const unsigned* ks[2] = {kpe, kse};
  const float* st[2] = {stpe, stse};
  float cf[2] = {0.25f * (*ag), 0.25f * (*bg)};

  float acc[8];
  for (int i = 0; i < 8; ++i) acc[i] = 0.f;
  for (int e = 0; e < 2; ++e) {
    for (int h = 0; h < HEADS; ++h) {
      v16bf A = load_frag(ks[e] + h * (HDP / 2), m0, lane, stride);
      v16bf B = load_frag(qs[e] + h * (HDP / 2), n0, lane, stride);
      v8f c = {};
      c = wmma_bf16(A, B, c);
      float M = st[e][(ncol * HEADS + h) * 2 + 0];
      float Si = st[e][(ncol * HEADS + h) * 2 + 1];
      for (int i = 0; i < 8; ++i) acc[i] += cf[e] * __expf(c[i] - M) * Si;
    }
  }
  for (int i = 0; i < 8; ++i) {
    int m = mrow0 + i;
    unsigned bits = bitsT[(size_t)m * (NN / 32) + (ncol >> 5)];
    float v = ((bits >> (ncol & 31)) & 1u) ? 0.f : acc[i];
    sampleT[(size_t)m * NN + ncol] = v;
  }
  float acc2[8];
  for (int i = 0; i < 8; ++i) acc2[i] = 0.f;
  for (int h = 0; h < HEADS; ++h) {
    v16bf A = load_frag(kae + h * (HDP / 2), m0, lane, stride);
    v16bf B = load_frag(qae + h * (HDP / 2), n0, lane, stride);
    v8f c = {};
    c = wmma_bf16(A, B, c);
    float M = stae[(ncol * HEADS + h) * 2 + 0];
    float Si = stae[(ncol * HEADS + h) * 2 + 1];
    for (int i = 0; i < 8; ++i) acc2[i] += 0.25f * __expf(c[i] - M) * Si;
  }
  for (int i = 0; i < 8; ++i)
    aegT[(size_t)(mrow0 + i) * NN + ncol] = acc2[i];
}

// ------------------------------ adjacency bitmask ---------------------------

__global__ void k_adj(const int* __restrict__ src, const int* __restrict__ dst,
                      unsigned* __restrict__ bitsT) {
  int e = blockIdx.x * blockDim.x + threadIdx.x;
  if (e >= EE) return;
  int s = src[e], d = dst[e];
  atomicOr(&bitsT[(size_t)d * (NN / 32) + (s >> 5)], 1u << (s & 31));
}

// ------------------------------ top-K + normalize + scatter -----------------
// One block per column m (= row of sampleT). K = 32.

__global__ void k_topk_scatter(const float* __restrict__ sampleT,
                               const float* __restrict__ aegT,
                               const float* __restrict__ aefeat,
                               const float* __restrict__ ag, const float* __restrict__ bg,
                               const float* __restrict__ cg,
                               float* __restrict__ out_global) {
  int m = blockIdx.x, t = threadIdx.x;          // 256 threads
  __shared__ float buf[NN];
  __shared__ float rv[256];
  __shared__ int   ri[256];
  __shared__ int   sidx[32];
  __shared__ float sval[32];
  __shared__ float colsum;
  const float* row = sampleT + (size_t)m * NN;
  for (int j = t; j < NN; j += 256) buf[j] = row[j];
  __syncthreads();
  for (int k = 0; k < 32; ++k) {
    float bv = -1e30f; int bi = 0;
    for (int j = t; j < NN; j += 256) {
      float v = buf[j];
      if (v > bv) { bv = v; bi = j; }
    }
    rv[t] = bv; ri[t] = bi;
    __syncthreads();
    for (int s = 128; s > 0; s >>= 1) {
      if (t < s && rv[t + s] > rv[t]) { rv[t] = rv[t + s]; ri[t] = ri[t + s]; }
      __syncthreads();
    }
    if (t == 0) { sidx[k] = ri[0]; sval[k] = rv[0]; buf[ri[0]] = -1e30f; }
    __syncthreads();
  }
  if (t < 32) {
    float g = 0.5f * ((*ag) + (*bg)) * sval[t] +
              (*cg) * aegT[(size_t)m * NN + sidx[t]];
    sval[t] = g;
  }
  __syncthreads();
  if (t == 0) {
    float s = 0.f;
    for (int k = 0; k < 32; ++k) s += sval[k];
    colsum = s;
  }
  __syncthreads();
  float inv = 1.f / colsum;
  for (int u = t; u < 32 * 64; u += 256) {
    int k = u >> 6, c = u & 63;
    atomicAdd(&out_global[sidx[k] * 64 + c], sval[k] * inv * aefeat[m * 64 + c]);
  }
}

// ------------------------------ final combine + LN --------------------------

__global__ void k_final(const float* __restrict__ out_local, const float* __restrict__ out_global,
                        const float* __restrict__ lw, const float* __restrict__ gw,
                        const float* __restrict__ lng, const float* __restrict__ lnb,
                        const float* __restrict__ x, float* __restrict__ out) {
  int n = blockIdx.x, t = threadIdx.x;          // 128 threads
  __shared__ float v[64];
  __shared__ float mu_s, rv_s;
  if (t < 64) {
    float acc = 0.f;
    for (int j = 0; j < 64; ++j)
      acc += out_local[n * 64 + j] * lw[t * 64 + j] +
             out_global[n * 64 + j] * gw[t * 64 + j];
    v[t] = acc;
  }
  __syncthreads();
  if (t == 0) {
    float mu = 0.f;
    for (int j = 0; j < 64; ++j) mu += v[j];
    mu *= (1.f / 64.f);
    float var = 0.f;
    for (int j = 0; j < 64; ++j) { float d = v[j] - mu; var += d * d; }
    var *= (1.f / 64.f);
    mu_s = mu; rv_s = rsqrtf(var + 1e-5f);
  }
  __syncthreads();
  if (t < 64)       out[n * 96 + t] = (v[t] - mu_s) * rv_s * lng[t] + lnb[t];
  else if (t < 96)  out[n * 96 + t] = x[n * 96 + t];   // pe0/se0 passthrough
}

// ------------------------------ host ---------------------------------------

extern "C" void kernel_launch(void* const* d_in, const int* in_sizes, int n_in,
                              void* d_out, int out_size, void* d_ws, size_t ws_size,
                              hipStream_t stream) {
  (void)in_sizes; (void)n_in; (void)out_size; (void)ws_size;
  const float* x = (const float*)d_in[0];
  const int* ei = (const int*)d_in[1];
  const int* src = ei;
  const int* dst = ei + EE;
  // param base indices: ae=3, pe=14, se=25 (w1,b1,w2,b2, gat_w,asrc,adst, wq,bq,wk,bk)
  const int AEB = 3, PEB = 14, SEB = 25;
  const float* gcn_w = (const float*)d_in[36];
  const float* gcn_b = (const float*)d_in[37];
  const float* a_l = (const float*)d_in[38];
  const float* b_l = (const float*)d_in[39];
  const float* c_l = (const float*)d_in[40];
  const float* a_g = (const float*)d_in[41];
  const float* b_g = (const float*)d_in[42];
  const float* c_g = (const float*)d_in[43];
  const float* local_w = (const float*)d_in[44];
  const float* global_w = (const float*)d_in[45];
  const float* ln_g = (const float*)d_in[46];
  const float* ln_b = (const float*)d_in[47];

  // workspace carve-up
  char* w = (char*)d_ws;
  size_t cur = 0;
  auto alloc = [&](size_t b) -> void* {
    void* p = w + cur;
    cur = (cur + b + 255) & ~(size_t)255;
    return p;
  };
  float* ae = (float*)alloc((size_t)NN * 64 * 4);
  float* pe = (float*)alloc((size_t)NN * 16 * 4);
  float* se = (float*)alloc((size_t)NN * 16 * 4);
  float* ssrc = (float*)alloc((size_t)NN * 4);
  float* sdst = (float*)alloc((size_t)NN * 4);
  float* tmp_a = (float*)alloc((size_t)EE * 4);
  unsigned* seg_m = (unsigned*)alloc((size_t)NN * 4);
  float* seg_s = (float*)alloc((size_t)NN * 4);
  float* local_attn = (float*)alloc((size_t)EE * 4);
  float* h_gcn = (float*)alloc((size_t)NN * 64 * 4);
  float* deg = (float*)alloc((size_t)NN * 4);
  float* out_local = (float*)alloc((size_t)NN * 64 * 4);
  unsigned short* qpad[3];
  unsigned short* kpad[3];
  float* stats[3];
  for (int e = 0; e < 3; ++e) {
    qpad[e] = (unsigned short*)alloc((size_t)NN * HEADS * HDP * 2);
    kpad[e] = (unsigned short*)alloc((size_t)NN * HEADS * HDP * 2);
    stats[e] = (float*)alloc((size_t)NN * HEADS * 2 * 4);
  }
  unsigned* bitsT = (unsigned*)alloc((size_t)NN * (NN / 32) * 4);
  float* sampleT = (float*)alloc((size_t)NN * NN * 4);
  float* aegT = (float*)alloc((size_t)NN * NN * 4);
  float* out_global = (float*)alloc((size_t)NN * 64 * 4);

  // ---- init ----
  for (int e = 0; e < 3; ++e) {
    hipMemsetAsync(qpad[e], 0, (size_t)NN * HEADS * HDP * 2, stream);
    hipMemsetAsync(kpad[e], 0, (size_t)NN * HEADS * HDP * 2, stream);
  }
  hipMemsetAsync(bitsT, 0, (size_t)NN * (NN / 32) * 4, stream);
  hipMemsetAsync(local_attn, 0, (size_t)EE * 4, stream);
  hipMemsetAsync(out_local, 0, (size_t)NN * 64 * 4, stream);
  hipMemsetAsync(out_global, 0, (size_t)NN * 64 * 4, stream);
  hipMemsetD32Async((hipDeviceptr_t)deg, 0x3F800000 /*1.0f*/, NN, stream);

  // ---- encoders ----
  k_mlp<64><<<NN, 128, 0, stream>>>(x, 0, (const float*)d_in[AEB], (const float*)d_in[AEB + 1],
                                    (const float*)d_in[AEB + 2], (const float*)d_in[AEB + 3], ae);
  k_mlp<16><<<NN, 32, 0, stream>>>(x, 64, (const float*)d_in[PEB], (const float*)d_in[PEB + 1],
                                   (const float*)d_in[PEB + 2], (const float*)d_in[PEB + 3], pe);
  k_mlp<16><<<NN, 32, 0, stream>>>(x, 80, (const float*)d_in[SEB], (const float*)d_in[SEB + 1],
                                   (const float*)d_in[SEB + 2], (const float*)d_in[SEB + 3], se);

  k_adj<<<(EE + 255) / 256, 256, 0, stream>>>(src, dst, bitsT);

  // ---- GAT local attention (pe->a_l, se->b_l, ae->c_l) ----
  struct { const float* feat; int base; const float* coef; int c; } gats[3] = {
      {pe, PEB, a_l, 16}, {se, SEB, b_l, 16}, {ae, AEB, c_l, 64}};
  for (int g = 0; g < 3; ++g) {
    hipMemsetD32Async((hipDeviceptr_t)seg_m, 0x007FFFFF /* key(-inf) */, NN, stream);
    hipMemsetAsync(seg_s, 0, (size_t)NN * 4, stream);
    const float* gw2 = (const float*)d_in[gats[g].base + 4];
    const float* as = (const float*)d_in[gats[g].base + 5];
    const float* ad = (const float*)d_in[gats[g].base + 6];
    if (gats[g].c == 16)
      k_gat_node<16><<<(NN + 127) / 128, 128, 0, stream>>>(gats[g].feat, gw2, as, ad, ssrc, sdst);
    else
      k_gat_node<64><<<(NN + 127) / 128, 128, 0, stream>>>(gats[g].feat, gw2, as, ad, ssrc, sdst);
    k_gat_edge_max<<<(EE + 255) / 256, 256, 0, stream>>>(src, dst, ssrc, sdst, tmp_a, seg_m);
    k_gat_edge_exp<<<(EE + 255) / 256, 256, 0, stream>>>(dst, tmp_a, seg_m, seg_s);
    k_gat_edge_alpha<<<(EE + 255) / 256, 256, 0, stream>>>(dst, tmp_a, seg_s, gats[g].coef, local_attn);
  }

  // ---- local GCN branch ----
  k_gcn_h<<<(NN * 64 + 255) / 256, 256, 0, stream>>>(ae, gcn_w, h_gcn);
  k_deg_scatter<<<(EE + 255) / 256, 256, 0, stream>>>(dst, local_attn, deg);
  k_dinv<<<(NN + 255) / 256, 256, 0, stream>>>(deg);
  k_local_scatter<<<(EE * 64) / 256, 256, 0, stream>>>(src, dst, local_attn, deg, h_gcn, out_local);
  k_local_self<<<(NN * 64) / 256, 256, 0, stream>>>(deg, h_gcn, gcn_b, out_local);

  // ---- MHA Q/K staging (bf16, K padded to 32) ----
  // enc order for pass2 arrays: 0=pe, 1=se, 2=ae
  k_qk<16><<<(NN * 16 + 255) / 256, 256, 0, stream>>>(
      pe, (const float*)d_in[PEB + 7], (const float*)d_in[PEB + 8],
      (const float*)d_in[PEB + 9], (const float*)d_in[PEB + 10], 0.5f, qpad[0], kpad[0]);
  k_qk<16><<<(NN * 16 + 255) / 256, 256, 0, stream>>>(
      se, (const float*)d_in[SEB + 7], (const float*)d_in[SEB + 8],
      (const float*)d_in[SEB + 9], (const float*)d_in[SEB + 10], 0.5f, qpad[1], kpad[1]);
  k_qk<64><<<(NN * 64 + 255) / 256, 256, 0, stream>>>(
      ae, (const float*)d_in[AEB + 7], (const float*)d_in[AEB + 8],
      (const float*)d_in[AEB + 9], (const float*)d_in[AEB + 10], 0.25f, qpad[2], kpad[2]);

  // ---- pass 1: per-row softmax stats (WMMA) ----
  dim3 wblk(32, 4);
  for (int e = 0; e < 3; ++e)
    k_mha_stats<<<NN / 16, wblk, 0, stream>>>((const unsigned*)qpad[e],
                                              (const unsigned*)kpad[e], stats[e]);

  // ---- pass 2: fused transposed probability matrices (WMMA) ----
  dim3 pgrid(NN / 16, NN / 64);
  k_pass2<<<pgrid, wblk, 0, stream>>>(
      (const unsigned*)qpad[0], (const unsigned*)kpad[0],
      (const unsigned*)qpad[1], (const unsigned*)kpad[1],
      (const unsigned*)qpad[2], (const unsigned*)kpad[2],
      stats[0], stats[1], stats[2], a_g, b_g, bitsT, sampleT, aegT);

  // ---- top-K per column + normalize + sparse g @ ae ----
  k_topk_scatter<<<NN, 256, 0, stream>>>(sampleT, aegT, ae, a_g, b_g, c_g, out_global);

  // ---- final projection + layernorm + passthrough ----
  k_final<<<NN, 128, 0, stream>>>(out_local, out_global, local_w, global_w,
                                  ln_g, ln_b, x, (float*)d_out);
}
